// E_GCL_67156108640471
// MI455X (gfx1250) — matrix operand order repaired and checked by reading, not compiled, not run on hardware
//
#include <hip/hip_runtime.h>
#include <stdint.h>
#include <stddef.h>

// ---------------- problem constants ----------------
constexpr int NN   = 20000;    // nodes
constexpr int NE   = 640000;   // edges
constexpr int DIM  = 128;      // feature dim

// workspace layout (bytes)
// [0, 229376): packed bf16 weight fragments (u32 view)
//   Wa  : u32[    0 .. 8192)   (We1[:, 0:128]^T)    32 frags
//   Wb  : u32[ 8192 ..16384)   (We1[:,128:256]^T)   32 frags
//   We2 : u32[16384 ..24576)                        32 frags
//   Wc1 : u32[24576 ..32768)                        32 frags
//   Wn1 : u32[32768 ..49152)   K=256                64 frags
//   Wn2 : u32[49152 ..57344)                        32 frags
// then: __bf16 hbf[N*128]; float agg_h[N*128]; float agg_c[N*3]; float cnt[N]
constexpr size_t WPACK_U32   = 57344;
constexpr size_t WPACK_BYTES = WPACK_U32 * 4;
constexpr size_t HBF_BYTES   = (size_t)NN * DIM * 2;

typedef __attribute__((ext_vector_type(16))) __bf16 v16bf;
typedef __attribute__((ext_vector_type(8)))  float  v8f;

union Frag16 { v16bf v; uint32_t u[8]; uint4 q[2]; };
union Pair   { __bf16 b[2]; uint32_t u; };

__device__ __forceinline__ float silu(float x) {
  return x * __builtin_amdgcn_rcpf(1.0f + __expf(-x));   // v_exp + v_rcp
}
// fire-and-forget f32 atomic add (no return -> STOREcnt path)
__device__ __forceinline__ void atomicAddF32(float* p, float v) {
  asm volatile("global_atomic_add_f32 %0, %1, off" : : "v"(p), "v"(v) : "memory");
}

// ---------------- weight packing ----------------
// B-fragment (32x16, bf16) lane layout:
//   lane L, VGPR j, half hh : n = nn*16 + (L&15), k = kk*32 + (L>>4)*8 + 2*j + (j>=4 ? 8 : 0) + hh
__global__ void __launch_bounds__(256) prep_pack(
    const float* __restrict__ We1, const float* __restrict__ We2,
    const float* __restrict__ Wc1, const float* __restrict__ Wn1,
    const float* __restrict__ Wn2, uint32_t* __restrict__ wpack) {
  int id = blockIdx.x * 256 + threadIdx.x;
  if (id >= (int)WPACK_U32) return;
  const float* W; int stride, coloff, base;
  if      (id <  8192) { W = We1; stride = 257; coloff = 0;   base = 0;     }
  else if (id < 16384) { W = We1; stride = 257; coloff = 128; base = 8192;  }
  else if (id < 24576) { W = We2; stride = 128; coloff = 0;   base = 16384; }
  else if (id < 32768) { W = Wc1; stride = 128; coloff = 0;   base = 24576; }
  else if (id < 49152) { W = Wn1; stride = 256; coloff = 0;   base = 32768; }
  else                 { W = Wn2; stride = 128; coloff = 0;   base = 49152; }
  int p    = id - base;
  int frag = p >> 8;            // 256 u32 per fragment
  int lane = (p >> 3) & 31;
  int j    = p & 7;
  int kk = frag >> 3, nn = frag & 7;
  int n  = nn * 16 + (lane & 15);
  int k0 = kk * 32 + (lane >> 4) * 8 + 2 * j + (j >= 4 ? 8 : 0);
  Pair pr;
  pr.b[0] = (__bf16)W[n * stride + coloff + k0];
  pr.b[1] = (__bf16)W[n * stride + coloff + k0 + 1];
  wpack[id] = pr.u;
}

// h (f32, row-major) -> hbf (bf16, row-major); 4 elements per thread
__global__ void __launch_bounds__(256) conv_h(const float* __restrict__ h,
                                              __bf16* __restrict__ hbf) {
  int id = blockIdx.x * 256 + threadIdx.x;          // 640000 threads
  if (id >= NN * DIM / 4) return;
  float4 f = *(const float4*)(h + (size_t)id * 4);
  union { __bf16 b[4]; uint2 u; } o;
  o.b[0] = (__bf16)f.x; o.b[1] = (__bf16)f.y;
  o.b[2] = (__bf16)f.z; o.b[3] = (__bf16)f.w;
  *(uint2*)(hbf + (size_t)id * 4) = o.u;
}

__global__ void zero_f32(float* __restrict__ p, size_t n) {
  size_t i = (size_t)blockIdx.x * blockDim.x + threadIdx.x;
  size_t st = (size_t)gridDim.x * blockDim.x;
  for (; i < n; i += st) p[i] = 0.0f;
}

// ---------------- edge kernel ----------------
// 8 waves/block, 16 edges per wave, 128 edges/block, 5000 blocks.
// Weights (Wa|Wb|We2|Wc1, 128 KB) staged in LDS; B frags read via ds_load_b128.
__global__ void __launch_bounds__(256, 1) egcl_edge(
    const __bf16* __restrict__ hbf, const float* __restrict__ coord,
    const int* __restrict__ eidx,
    const float* __restrict__ We1, const float* __restrict__ be1,
    const float* __restrict__ be2, const float* __restrict__ bc1,
    const float* __restrict__ Wc2,
    const uint32_t* __restrict__ wpack,
    float* __restrict__ agg_h, float* __restrict__ agg_c,
    float* __restrict__ cnt) {
  __shared__ uint4  wlds[8192];           // 131072 B: 4 matrices x 32 frags x 64 uint4
  __shared__ __bf16 staged[8][16][136];   // padded stride (bank-conflict free)
  __shared__ float  radial_s[8][16];
  __shared__ int    row_s[8][16];
  __shared__ float  coef_s[8][16];

  const int tid  = threadIdx.x;
  const int w    = tid >> 5;
  const int lane = tid & 31;
  const int m    = lane & 15;
  const int hi   = lane >> 4;
  const int eb   = blockIdx.x * 128 + w * 16;
  const int e    = eb + m;

  // cooperative copy of all 4 weight matrices into LDS
  {
    const uint4* wg = (const uint4*)wpack;
    #pragma unroll
    for (int i = 0; i < 32; ++i) wlds[tid + i * 256] = wg[tid + i * 256];
  }

  const int r = eidx[e];
  const int c = eidx[NE + e];
  const float dx = coord[r * 3 + 0] - coord[c * 3 + 0];
  const float dy = coord[r * 3 + 1] - coord[c * 3 + 1];
  const float dz = coord[r * 3 + 2] - coord[c * 3 + 2];
  const float radial = dx * dx + dy * dy + dz * dz;
  if (hi == 0) { radial_s[w][m] = radial; row_s[w][m] = r; }
  __syncthreads();

  // ---- layer 1: silu(h[r]@Wa + h[c]@Wb + radial*wr + be1) ----
  v8f acc[8] = {};
  #pragma unroll
  for (int pass = 0; pass < 2; ++pass) {
    const int src = (pass == 0) ? r : c;
    const int wbase = (pass == 0 ? 0 : 2048);      // uint4 units
    const __bf16* hrow = hbf + (size_t)src * DIM;
    #pragma unroll
    for (int kk = 0; kk < 4; ++kk) {
      Frag16 a;
      const int k0 = kk * 32 + hi * 8;
      a.q[0] = *(const uint4*)(hrow + k0);        // k0..k0+7
      a.q[1] = *(const uint4*)(hrow + k0 + 16);   // k0+16..k0+23
      #pragma unroll
      for (int nn = 0; nn < 8; ++nn) {
        Frag16 b;
        const int fo = wbase + (kk * 8 + nn) * 64 + lane * 2;
        b.q[0] = wlds[fo]; b.q[1] = wlds[fo + 1];
        acc[nn] = __builtin_amdgcn_wmma_f32_16x16x32_bf16(
            false, a.v, false, b.v, (short)0, acc[nn], false, false);
      }
    }
  }
  #pragma unroll
  for (int nn = 0; nn < 8; ++nn) {
    const int n = nn * 16 + m;
    const float bias = be1[n];
    const float wr = We1[n * 257 + 256];
    #pragma unroll
    for (int j = 0; j < 8; ++j) {
      const int mm = j + 8 * hi;
      staged[w][mm][n] = (__bf16)silu(acc[nn][j] + bias + radial_s[w][mm] * wr);
    }
  }

  // ---- layer 2: edge_feat = silu(ef@We2 + be2) ----
  #pragma unroll
  for (int nn = 0; nn < 8; ++nn) acc[nn] = (v8f){};
  #pragma unroll
  for (int kk = 0; kk < 4; ++kk) {
    Frag16 a;
    const int k0 = kk * 32 + hi * 8;
    const __bf16* sp = &staged[w][m][0];
    a.q[0] = *(const uint4*)(sp + k0);
    a.q[1] = *(const uint4*)(sp + k0 + 16);
    #pragma unroll
    for (int nn = 0; nn < 8; ++nn) {
      Frag16 b;
      const int fo = 4096 + (kk * 8 + nn) * 64 + lane * 2;
      b.q[0] = wlds[fo]; b.q[1] = wlds[fo + 1];
      acc[nn] = __builtin_amdgcn_wmma_f32_16x16x32_bf16(
          false, a.v, false, b.v, (short)0, acc[nn], false, false);
    }
  }
  // epilogue: scatter agg_h, restage edge_feat for Wc1
  #pragma unroll
  for (int nn = 0; nn < 8; ++nn) {
    const int n = nn * 16 + m;
    const float bias = be2[n];
    #pragma unroll
    for (int j = 0; j < 8; ++j) {
      const int mm = j + 8 * hi;
      const float y = silu(acc[nn][j] + bias);
      atomicAddF32(agg_h + (size_t)row_s[w][mm] * DIM + n, y);
      staged[w][mm][n] = (__bf16)y;
    }
  }

  // ---- layer 3: coef = silu(edge_feat@Wc1 + bc1) @ Wc2 ----
  #pragma unroll
  for (int nn = 0; nn < 8; ++nn) acc[nn] = (v8f){};
  #pragma unroll
  for (int kk = 0; kk < 4; ++kk) {
    Frag16 a;
    const int k0 = kk * 32 + hi * 8;
    const __bf16* sp = &staged[w][m][0];
    a.q[0] = *(const uint4*)(sp + k0);
    a.q[1] = *(const uint4*)(sp + k0 + 16);
    #pragma unroll
    for (int nn = 0; nn < 8; ++nn) {
      Frag16 b;
      const int fo = 6144 + (kk * 8 + nn) * 64 + lane * 2;
      b.q[0] = wlds[fo]; b.q[1] = wlds[fo + 1];
      acc[nn] = __builtin_amdgcn_wmma_f32_16x16x32_bf16(
          false, a.v, false, b.v, (short)0, acc[nn], false, false);
    }
  }
  float s[8] = {};
  #pragma unroll
  for (int nn = 0; nn < 8; ++nn) {
    const int n = nn * 16 + m;
    const float bias = bc1[n];
    const float wv = Wc2[n];
    #pragma unroll
    for (int j = 0; j < 8; ++j) s[j] += silu(acc[nn][j] + bias) * wv;
  }
  #pragma unroll
  for (int off = 1; off < 16; off <<= 1) {
    #pragma unroll
    for (int j = 0; j < 8; ++j) s[j] += __shfl_xor(s[j], off, 32);
  }
  if (m == 0) {   // lanes 0 and 16 hold coefs for m=0..7 / 8..15
    #pragma unroll
    for (int j = 0; j < 8; ++j) coef_s[w][j + 8 * hi] = s[j];
  }
  __syncthreads();
  if (hi == 0) {
    const float coef = coef_s[w][m];
    atomicAddF32(agg_c + (size_t)r * 3 + 0, dx * coef);
    atomicAddF32(agg_c + (size_t)r * 3 + 1, dy * coef);
    atomicAddF32(agg_c + (size_t)r * 3 + 2, dz * coef);
    atomicAddF32(cnt + r, 1.0f);
  }
}

// ---------------- node kernel ----------------
__global__ void __launch_bounds__(256, 1) egcl_node(
    const float* __restrict__ h, const __bf16* __restrict__ hbf,
    const float* __restrict__ coord,
    const float* __restrict__ bn1, const float* __restrict__ bn2,
    const uint32_t* __restrict__ wpack,
    const float* __restrict__ agg_h, const float* __restrict__ agg_c,
    const float* __restrict__ cnt, float* __restrict__ out) {
  __shared__ uint4  wlds[6144];           // 98304 B: Wn1 (64 frags) + Wn2 (32 frags)
  __shared__ __bf16 staged[8][16][136];

  const int tid  = threadIdx.x;
  const int w    = tid >> 5;
  const int lane = tid & 31;
  const int m    = lane & 15;
  const int hi   = lane >> 4;
  const int nb   = blockIdx.x * 128 + w * 16;
  const int nd   = nb + m;
  const int ndc  = nd < NN ? nd : NN - 1;

  {
    const uint4* wg = (const uint4*)(wpack + 32768);
    #pragma unroll
    for (int i = 0; i < 24; ++i) wlds[tid + i * 256] = wg[tid + i * 256];
  }
  __syncthreads();

  // ---- layer 1: silu([h | agg_h] @ Wn1 + bn1), K = 256 ----
  v8f acc[8] = {};
  #pragma unroll
  for (int kk = 0; kk < 8; ++kk) {
    Frag16 a;
    const int k0 = hi * 8;
    if (kk < 4) {
      const __bf16* src = hbf + (size_t)ndc * DIM + kk * 32;
      a.q[0] = *(const uint4*)(src + k0);
      a.q[1] = *(const uint4*)(src + k0 + 16);
    } else {
      const float* src = agg_h + (size_t)ndc * DIM + (kk - 4) * 32;
      float4 f0 = *(const float4*)(src + k0);
      float4 f1 = *(const float4*)(src + k0 + 4);
      float4 f2 = *(const float4*)(src + k0 + 16);
      float4 f3 = *(const float4*)(src + k0 + 20);
      a.v[0]  = (__bf16)f0.x; a.v[1]  = (__bf16)f0.y;
      a.v[2]  = (__bf16)f0.z; a.v[3]  = (__bf16)f0.w;
      a.v[4]  = (__bf16)f1.x; a.v[5]  = (__bf16)f1.y;
      a.v[6]  = (__bf16)f1.z; a.v[7]  = (__bf16)f1.w;
      a.v[8]  = (__bf16)f2.x; a.v[9]  = (__bf16)f2.y;
      a.v[10] = (__bf16)f2.z; a.v[11] = (__bf16)f2.w;
      a.v[12] = (__bf16)f3.x; a.v[13] = (__bf16)f3.y;
      a.v[14] = (__bf16)f3.z; a.v[15] = (__bf16)f3.w;
    }
    #pragma unroll
    for (int nn = 0; nn < 8; ++nn) {
      Frag16 b;
      const int fo = (kk * 8 + nn) * 64 + lane * 2;
      b.q[0] = wlds[fo]; b.q[1] = wlds[fo + 1];
      acc[nn] = __builtin_amdgcn_wmma_f32_16x16x32_bf16(
          false, a.v, false, b.v, (short)0, acc[nn], false, false);
    }
  }
  #pragma unroll
  for (int nn = 0; nn < 8; ++nn) {
    const int n = nn * 16 + m;
    const float bias = bn1[n];
    #pragma unroll
    for (int j = 0; j < 8; ++j) {
      const int mm = j + 8 * hi;
      staged[w][mm][n] = (__bf16)silu(acc[nn][j] + bias);
    }
  }

  // ---- layer 2: h_out = h + (t @ Wn2 + bn2) ----
  #pragma unroll
  for (int nn = 0; nn < 8; ++nn) acc[nn] = (v8f){};
  #pragma unroll
  for (int kk = 0; kk < 4; ++kk) {
    Frag16 a;
    const int k0 = kk * 32 + hi * 8;
    const __bf16* sp = &staged[w][m][0];
    a.q[0] = *(const uint4*)(sp + k0);
    a.q[1] = *(const uint4*)(sp + k0 + 16);
    #pragma unroll
    for (int nn = 0; nn < 8; ++nn) {
      Frag16 b;
      const int fo = 4096 + (kk * 8 + nn) * 64 + lane * 2;
      b.q[0] = wlds[fo]; b.q[1] = wlds[fo + 1];
      acc[nn] = __builtin_amdgcn_wmma_f32_16x16x32_bf16(
          false, a.v, false, b.v, (short)0, acc[nn], false, false);
    }
  }
  #pragma unroll
  for (int nn = 0; nn < 8; ++nn) {
    const int n = nn * 16 + m;
    const float bias = bn2[n];
    #pragma unroll
    for (int j = 0; j < 8; ++j) {
      const int mm = j + 8 * hi;
      const int node = nb + mm;
      if (node < NN)
        out[(size_t)node * DIM + n] = acc[nn][j] + bias + h[(size_t)node * DIM + n];
    }
  }

  // ---- coord epilogue: coord + agg_c / max(cnt,1) ----
  if (hi == 0 && nd < NN) {
    const float cc = cnt[nd];
    const float den = cc > 1.0f ? cc : 1.0f;
    float* co = out + (size_t)NN * DIM;
    #pragma unroll
    for (int i = 0; i < 3; ++i)
      co[(size_t)nd * 3 + i] = coord[(size_t)nd * 3 + i] + agg_c[(size_t)nd * 3 + i] / den;
  }
}

// ---------------- launch ----------------
extern "C" void kernel_launch(void* const* d_in, const int* in_sizes, int n_in,
                              void* d_out, int out_size, void* d_ws, size_t ws_size,
                              hipStream_t stream) {
  const float* h     = (const float*)d_in[0];
  const float* coord = (const float*)d_in[1];
  const int*   eidx  = (const int*)  d_in[2];
  const float* We1   = (const float*)d_in[3];
  const float* be1   = (const float*)d_in[4];
  const float* We2   = (const float*)d_in[5];
  const float* be2   = (const float*)d_in[6];
  const float* Wn1   = (const float*)d_in[7];
  const float* bn1   = (const float*)d_in[8];
  const float* Wn2   = (const float*)d_in[9];
  const float* bn2   = (const float*)d_in[10];
  const float* Wc1   = (const float*)d_in[11];
  const float* bc1   = (const float*)d_in[12];
  const float* Wc2   = (const float*)d_in[13];

  uint32_t* wpack = (uint32_t*)d_ws;
  __bf16*   hbf   = (__bf16*)((char*)d_ws + WPACK_BYTES);
  float* agg_h = (float*)((char*)d_ws + WPACK_BYTES + HBF_BYTES);
  float* agg_c = agg_h + (size_t)NN * DIM;
  float* cnt   = agg_c + (size_t)NN * 3;
  float* out   = (float*)d_out;

  prep_pack<<<(WPACK_U32 + 255) / 256, 256, 0, stream>>>(We1, We2, Wc1, Wn1, Wn2, wpack);
  conv_h<<<(NN * DIM / 4 + 255) / 256, 256, 0, stream>>>(h, hbf);
  zero_f32<<<1024, 256, 0, stream>>>(agg_h, (size_t)NN * (DIM + 3 + 1));
  egcl_edge<<<NE / 128, 256, 0, stream>>>(hbf, coord, eidx, We1, be1, be2, bc1, Wc2,
                                          wpack, agg_h, agg_c, cnt);
  egcl_node<<<(NN + 127) / 128, 256, 0, stream>>>(h, hbf, coord, bn1, bn2, wpack,
                                                  agg_h, agg_c, cnt, out);
}
